// LSTM_48172353192177
// MI455X (gfx1250) — matrix operand order repaired and checked by reading, not compiled, not run on hardware
//
#include <hip/hip_runtime.h>
#include <hip/hip_bf16.h>

typedef __attribute__((ext_vector_type(16))) __bf16 v16bf;
typedef __attribute__((ext_vector_type(8)))  float  v8f;

#define E_   1024
#define S_   128
#define B_   64
#define K2E  2048
#define HSZ  (B_ * E_)              // one [64,1024] state
#define KC   64                     // K-chunk staged in LDS per stage
#define NSTG (K2E / KC)             // 32 stages
#define APAD 72                     // 64 + 8 elements: conflict-free LDS rows

union Frag16 { v16bf v; uint4 q[2]; };

__device__ __forceinline__ float sigf(float x) {
    return 1.0f / (1.0f + __expf(-x));
}

// ---------------------------------------------------------------------------
// x: [B,S,E] f32 -> [S,B,E] bf16
// ---------------------------------------------------------------------------
__global__ void xconv_kernel(const float* __restrict__ in, __bf16* __restrict__ out) {
    size_t i = (size_t)blockIdx.x * blockDim.x + threadIdx.x;
    if (i >= (size_t)S_ * B_ * E_) return;
    int t = (int)(i / (B_ * E_));
    int r = (int)(i % (B_ * E_));
    int b = r / E_, e = r % E_;
    out[i] = (__bf16)in[((size_t)b * S_ + t) * E_ + e];
}

// ---------------------------------------------------------------------------
// W [2E,E] f32 row-major (k,n) -> Wt [E,2E] bf16 (n,k); LDS-tiled transpose.
// ---------------------------------------------------------------------------
__global__ void wtrans_kernel(const float* __restrict__ src, __bf16* __restrict__ dst) {
    __shared__ float tile[32][33];
    int n0 = blockIdx.x * 32, k0 = blockIdx.y * 32;
    int tx = threadIdx.x, ty = threadIdx.y;   // block (32,8)
#pragma unroll
    for (int j = 0; j < 32; j += 8)
        tile[ty + j][tx] = src[(size_t)(k0 + ty + j) * E_ + n0 + tx];
    __syncthreads();
#pragma unroll
    for (int j = 0; j < 32; j += 8)
        dst[(size_t)(n0 + ty + j) * K2E + k0 + tx] = (__bf16)tile[tx][ty + j];
}

// ---------------------------------------------------------------------------
// Initialize h (parity 0, both dirs/layers) bf16 and c f32 from inputs.
// ---------------------------------------------------------------------------
__global__ void initst_kernel(const float* __restrict__ hid, const float* __restrict__ cell,
                              __bf16* __restrict__ hbuf, float* __restrict__ cbuf) {
    size_t i = (size_t)blockIdx.x * blockDim.x + threadIdx.x;
    if (i >= (size_t)4 * HSZ) return;        // 4 = dir*layer states
    int be = (int)(i % HSZ);                 // hid/cell are [B,E] == flat be
    hbuf[i] = (__bf16)hid[be];               // parity-0 region of hbuf
    cbuf[i] = cell[be];
}

__global__ void initb_kernel(const float* ff, const float* fi, const float* fc,
                             const float* bfp, const float* bip, const float* bcp,
                             float* __restrict__ bias) {
    int i = blockIdx.x * blockDim.x + threadIdx.x;
    if (i >= 12 * E_) return;
    int mat = i >> 10, n = i & (E_ - 1);
    int dir = mat / 6, rem = mat % 6, layer = rem / 3, gate = rem % 3;
    const float* src = (dir == 0) ? (gate == 0 ? ff : (gate == 1 ? fi : fc))
                                  : (gate == 0 ? bfp : (gate == 1 ? bip : bcp));
    bias[i] = src[layer * E_ + n];
}

// ---------------------------------------------------------------------------
// One LSTM step for one layer, both directions (blockIdx.y = dir).
// z = [h_layer | inp]; 3 gate GEMMs [64,2048]x[2048,1024] via bf16 WMMA.
// Block = 8 waves = M64 x N32 x 3 gates slab; A/B staged through
// double-buffered LDS so B weights are read from L2 exactly once.
// grid (32, 2), block 256. Wave (m,n) tile: m = wave&3, n = wave>>2.
// hbuf layout: [parity][dir][layer][64][1024] bf16 (ping-pong on t parity).
// wt layout:   [(dir*2+layer)*3+gate][n=1024][k=2048] bf16.
// ---------------------------------------------------------------------------
__global__ void lstm_step_kernel(const __bf16* __restrict__ xbf,
                                 const __bf16* __restrict__ wt,
                                 const float* __restrict__ bias,
                                 __bf16* __restrict__ hbuf,
                                 float* __restrict__ cbuf,
                                 __bf16* __restrict__ hist,
                                 int layer, int t) {
    __shared__ __align__(16) __bf16 As[2][B_][APAD];      // 18 KB
    __shared__ __align__(16) __bf16 Bs[2][3][32][APAD];   // 27 KB

    const int dir = blockIdx.y;
    const int p   = t & 1;
    const int tD  = (dir == 0) ? t : (S_ - 1 - t);

    const __bf16* A1  = hbuf + (size_t)(((p      * 2 + dir) * 2) + layer) * HSZ;
    __bf16*       hO  = hbuf + (size_t)((((1 - p) * 2 + dir) * 2) + layer) * HSZ;
    const __bf16* A2  = (layer == 0)
        ? (xbf + (size_t)tD * HSZ)
        : (hbuf + (size_t)((((1 - p) * 2 + dir) * 2) + 0) * HSZ);   // h0 written this step
    const __bf16* W0  = wt   + (size_t)((dir * 2 + layer) * 3) * E_ * K2E;
    const float*  bs  = bias + (size_t)((dir * 2 + layer) * 3) * E_;
    float*        c   = cbuf + (size_t)(dir * 2 + layer) * HSZ;
    __bf16*       hh  = (layer == 1) ? (hist + ((size_t)dir * S_ + tD) * HSZ) : nullptr;

    const int tid   = threadIdx.x;
    const int lane  = tid & 31;
    const int wave  = tid >> 5;
    const int m0    = (wave & 3) << 4;       // M-tile base (0..48)
    const int nb    = (wave >> 2) << 4;      // N sub-tile base in slab (0 or 16)
    const int n0blk = blockIdx.x * 32;       // slab N base

    const int r16   = lane & 15;             // A row / B col within 16
    const int khalf = lane >> 4;
    const int akb   = khalf * 16;            // A frag: 16 contiguous at akb
    const int bkb   = khalf * 16;            // B frag: 16 contiguous at bkb
    const int ncol  = n0blk + nb + r16;

    // Staging assignments (per thread): A = 2 x 16B chunks, B = 3 x 16B chunks
    int arow[2], acol[2], bgat[3], bcol[3], bcof[3];
#pragma unroll
    for (int j = 0; j < 2; ++j) {
        int cid = tid + j * 256;             // 0..511 over 64 rows x 8 chunks
        arow[j] = cid >> 3;
        acol[j] = (cid & 7) * 8;
    }
#pragma unroll
    for (int j = 0; j < 3; ++j) {
        int cid = tid + j * 256;             // 0..767 over 3 gates x 32 cols x 8 chunks
        bgat[j] = cid >> 8;
        int r   = cid & 255;
        bcol[j] = r >> 3;
        bcof[j] = (r & 7) * 8;
    }

    uint4 pa[2], pb[3];
    auto gload = [&](int s) {
        const __bf16* asrc = ((s * KC < E_) ? A1 : A2) + ((s * KC) & (E_ - 1));
#pragma unroll
        for (int j = 0; j < 2; ++j)
            pa[j] = *(const uint4*)(asrc + (size_t)arow[j] * E_ + acol[j]);
#pragma unroll
        for (int j = 0; j < 3; ++j)
            pb[j] = *(const uint4*)(W0 + (size_t)bgat[j] * E_ * K2E
                                       + (size_t)(n0blk + bcol[j]) * K2E
                                       + s * KC + bcof[j]);
    };
    auto sstore = [&](int buf) {
#pragma unroll
        for (int j = 0; j < 2; ++j)
            *(uint4*)&As[buf][arow[j]][acol[j]] = pa[j];
#pragma unroll
        for (int j = 0; j < 3; ++j)
            *(uint4*)&Bs[buf][bgat[j]][bcol[j]][bcof[j]] = pb[j];
    };

    v8f accF = {}, accI = {}, accG = {};

    gload(0);
    sstore(0);
    __syncthreads();

#pragma unroll 2
    for (int s = 0; s < NSTG; ++s) {
        const int buf = s & 1;
        if (s + 1 < NSTG) gload(s + 1);      // overlap next stage with WMMAs

#pragma unroll
        for (int ks = 0; ks < KC; ks += 32) {
            Frag16 a;
            a.q[0] = *(const uint4*)&As[buf][m0 + r16][ks + akb];
            a.q[1] = *(const uint4*)&As[buf][m0 + r16][ks + akb + 8];
            Frag16 b0, b1, b2;
            b0.q[0] = *(const uint4*)&Bs[buf][0][nb + r16][ks + bkb];
            b0.q[1] = *(const uint4*)&Bs[buf][0][nb + r16][ks + bkb + 8];
            b1.q[0] = *(const uint4*)&Bs[buf][1][nb + r16][ks + bkb];
            b1.q[1] = *(const uint4*)&Bs[buf][1][nb + r16][ks + bkb + 8];
            b2.q[0] = *(const uint4*)&Bs[buf][2][nb + r16][ks + bkb];
            b2.q[1] = *(const uint4*)&Bs[buf][2][nb + r16][ks + bkb + 8];
            accF = __builtin_amdgcn_wmma_f32_16x16x32_bf16(false, a.v, false, b0.v, (short)0, accF, false, false);
            accI = __builtin_amdgcn_wmma_f32_16x16x32_bf16(false, a.v, false, b1.v, (short)0, accI, false, false);
            accG = __builtin_amdgcn_wmma_f32_16x16x32_bf16(false, a.v, false, b2.v, (short)0, accG, false, false);
        }

        if (s + 1 < NSTG) sstore((s + 1) & 1);
        __syncthreads();
    }

    const float biasF = bs[ncol];
    const float biasI = bs[E_ + ncol];
    const float biasG = bs[2 * E_ + ncol];
    const int   mbase = m0 + khalf * 8;

#pragma unroll
    for (int r = 0; r < 8; ++r) {
        const int m = mbase + r;
        float f  = sigf(accF[r] + biasF);
        float i  = sigf(accI[r] + biasI);
        float g  = tanhf(accG[r] + biasG);
        size_t idx = (size_t)m * E_ + ncol;
        float cn = c[idx] * f + i * g;
        c[idx]   = cn;
        float hn = tanhf(cn) * f;            // forget gate reused as output gate
        hO[idx]  = (__bf16)hn;
        if (hh) hh[idx] = (__bf16)hn;
    }
}

// ---------------------------------------------------------------------------
// w[b,t] = sum_e hist_fwd[t,b,e]*wordW[e] + hist_bwd[t,b,e]*wordW[E+e] + word_b
// one wave per (b,t); wave32 shuffle reduction.
// ---------------------------------------------------------------------------
__global__ void word_kernel(const __bf16* __restrict__ hist,
                            const float* __restrict__ wW, const float* __restrict__ wb,
                            float* __restrict__ wout) {
    int gw   = blockIdx.x * 8 + (threadIdx.x >> 5);
    int lane = threadIdx.x & 31;
    int b = gw >> 7, t = gw & (S_ - 1);
    const __bf16* hf = hist + (size_t)t * HSZ + (size_t)b * E_;
    const __bf16* hb = hist + ((size_t)S_ + t) * HSZ + (size_t)b * E_;
    float s = 0.0f;
    for (int e = lane; e < E_; e += 32)
        s += (float)hf[e] * wW[e] + (float)hb[e] * wW[E_ + e];
#pragma unroll
    for (int o = 16; o > 0; o >>= 1) s += __shfl_xor(s, o, 32);
    if (lane == 0) wout[b * S_ + t] = s + wb[0];
}

__global__ void pred_kernel(const float* __restrict__ w, const float* __restrict__ pW,
                            const float* __restrict__ pb, float* __restrict__ out) {
    int i = threadIdx.x;
    if (i >= B_ * 5) return;
    int b = i / 5, l = i % 5;
    float acc = pb[l];
    for (int t = 0; t < S_; ++t) acc += w[b * S_ + t] * pW[t * 5 + l];
    out[i] = acc;
}

// ---------------------------------------------------------------------------
extern "C" void kernel_launch(void* const* d_in, const int* in_sizes, int n_in,
                              void* d_out, int out_size, void* d_ws, size_t ws_size,
                              hipStream_t stream) {
    (void)in_sizes; (void)n_in; (void)out_size; (void)ws_size;

    const float* inputs  = (const float*)d_in[0];
    const float* cell    = (const float*)d_in[1];
    const float* hidden  = (const float*)d_in[2];
    // d_in[3] back_states, d_in[4] back_hidden: unused by the reference
    const float* fWf = (const float*)d_in[5];  const float* fbf = (const float*)d_in[6];
    const float* fWi = (const float*)d_in[7];  const float* fbi = (const float*)d_in[8];
    const float* fWc = (const float*)d_in[9];  const float* fbc = (const float*)d_in[10];
    const float* bWf = (const float*)d_in[11]; const float* bbf = (const float*)d_in[12];
    const float* bWi = (const float*)d_in[13]; const float* bbi = (const float*)d_in[14];
    const float* bWc = (const float*)d_in[15]; const float* bbc = (const float*)d_in[16];
    const float* word_W = (const float*)d_in[17];
    const float* word_b = (const float*)d_in[18];
    const float* pred_W = (const float*)d_in[19];
    const float* pred_b = (const float*)d_in[20];
    float* out = (float*)d_out;

    char* ws = (char*)d_ws;
    size_t off = 0;
    __bf16* xbf  = (__bf16*)(ws + off); off += (size_t)S_ * B_ * E_ * 2;          // 16 MB
    __bf16* wt   = (__bf16*)(ws + off); off += (size_t)12 * E_ * K2E * 2;          // 48 MB
    __bf16* hbuf = (__bf16*)(ws + off); off += (size_t)2 * 2 * 2 * HSZ * 2;        // 1 MB
    float*  cbuf = (float*) (ws + off); off += (size_t)2 * 2 * HSZ * 4;            // 1 MB
    __bf16* hist = (__bf16*)(ws + off); off += (size_t)2 * S_ * HSZ * 2;           // 32 MB
    float*  bias = (float*) (ws + off); off += (size_t)12 * E_ * 4;                // 48 KB
    float*  wout = (float*) (ws + off); off += (size_t)B_ * S_ * 4;                // 32 KB

    // Phase 0: conversions / init
    {
        size_t n = (size_t)S_ * B_ * E_;
        xconv_kernel<<<(unsigned)((n + 255) / 256), 256, 0, stream>>>(inputs, xbf);
    }
    const float* gw[2][3] = {{fWf, fWi, fWc}, {bWf, bWi, bWc}};
    for (int d = 0; d < 2; ++d)
        for (int l = 0; l < 2; ++l)
            for (int g = 0; g < 3; ++g) {
                int mat = (d * 2 + l) * 3 + g;
                wtrans_kernel<<<dim3(32, 64), dim3(32, 8), 0, stream>>>(
                    gw[d][g] + (size_t)l * K2E * E_, wt + (size_t)mat * E_ * K2E);
            }
    initst_kernel<<<(4 * HSZ + 255) / 256, 256, 0, stream>>>(hidden, cell, hbuf, cbuf);
    initb_kernel<<<(12 * E_ + 255) / 256, 256, 0, stream>>>(fbf, fbi, fbc, bbf, bbi, bbc, bias);

    // Phase 1: sequential recurrence (both directions per launch)
    for (int t = 0; t < S_; ++t) {
        lstm_step_kernel<<<dim3(32, 2), 256, 0, stream>>>(xbf, wt, bias, hbuf, cbuf, hist, 0, t);
        lstm_step_kernel<<<dim3(32, 2), 256, 0, stream>>>(xbf, wt, bias, hbuf, cbuf, hist, 1, t);
    }

    // Phase 2: word projection + prediction
    word_kernel<<<(B_ * S_) / 8, 256, 0, stream>>>(hist, word_W, word_b, wout);
    pred_kernel<<<1, 320, 0, stream>>>(wout, pred_W, pred_b, out);
}
